// TemporalGraphTower_68444598829206
// MI455X (gfx1250) — compile-verified, matrix operand order checked by reading, not run on hardware
//
#include <hip/hip_runtime.h>

#define B_EDGES 8192
#define NODE_DIM 128
#define EDGE_DIM 64
#define TIME_DIM 32
#define HIDDEN 256
#define IN_DIM 352
#define KNBR 20

typedef __attribute__((ext_vector_type(16))) _Float16 v16h;
typedef __attribute__((ext_vector_type(8)))  _Float16 v8h;
typedef __attribute__((ext_vector_type(8)))  float    v8f;

// ---------------------------------------------------------------- WMMA helpers

__device__ __forceinline__ v8f wmma16x16x32(v16h a, v16h b, v8f c) {
  // D = A(16x32 f16) * B(32x16 f16) + C(16x16 f32)
  return __builtin_amdgcn_wmma_f32_16x16x32_f16(false, a, false, b, (short)0, c,
                                                false, false);
}

// A-fragment from a row-major f16 matrix (LDS): lane m = lane&15, K chunks
// {k0 + (lane>>4)*8 .. +7} and {k0+16+(lane>>4)*8 .. +7}  (ISA 16-bit A layout)
__device__ __forceinline__ v16h load_a_rowmajor(const _Float16* base, int row0,
                                                int stride, int k0, int lane) {
  const _Float16* p =
      base + (size_t)(row0 + (lane & 15)) * stride + k0 + ((lane >> 4) << 3);
  v8h lo = *(const v8h*)p;
  v8h hi = *(const v8h*)(p + 16);
  v16h r;
#pragma unroll
  for (int i = 0; i < 8; ++i) { r[i] = lo[i]; r[i + 8] = hi[i]; }
  return r;
}

// B-fragment from a pre-packed weight buffer: 16 contiguous halves per lane.
__device__ __forceinline__ v16h load_b_pack(const _Float16* pack, int nt, int ks,
                                            int ksteps, int lane) {
  const _Float16* p = pack + ((((size_t)nt * ksteps + ks) * 32 + lane) << 4);
  return *(const v16h*)p;
}

// Epilogue: bias + ReLU -> f16 row-major LDS tile.
__device__ __forceinline__ void store_tile_relu(_Float16* dst, int stride,
                                                v8f acc, int mt, int nt,
                                                const float* bias, int lane) {
  int col = (nt << 4) + (lane & 15);
  float b = bias[col];
  int rb = (mt << 4) + ((lane >> 4) << 3);
#pragma unroll
  for (int i = 0; i < 8; ++i) {
    float v = acc[i] + b;
    dst[(rb + i) * stride + col] = (_Float16)(v > 0.f ? v : 0.f);
  }
}

// ---------------------------------------------------------------- weight prepack
// dst[((nt*ksteps+ks)*32+lane)*16 + j] holds W[kk][col] in f16 where
// col = nt*16 + (lane&15), kk = ks*32 + interleaved chunk (matches fragments).
__global__ void prepack_kernel(const float* __restrict__ src,
                               _Float16* __restrict__ dst, int Kdim, int N,
                               int transposed) {
  int id = blockIdx.x * blockDim.x + threadIdx.x;
  if (id >= Kdim * N) return;
  int j = id & 15;
  int lane = (id >> 4) & 31;
  int rest = id >> 9;
  int ksteps = Kdim >> 5;
  int ks = rest % ksteps;
  int nt = rest / ksteps;
  int col = (nt << 4) + (lane & 15);
  int kh = (lane >> 4) << 3;
  int kk = (ks << 5) + ((j < 8) ? (kh + j) : (16 + kh + (j - 8)));
  float v = transposed ? src[(size_t)col * Kdim + kk] : src[(size_t)kk * N + col];
  dst[id] = (_Float16)v;
}

// ---------------------------------------------------------------- edge init
__global__ void init_edges_kernel(
    const int* __restrict__ src_nodes, const int* __restrict__ dst_nodes,
    const float* __restrict__ edge_attrs, const float* __restrict__ timestamps,
    const float* __restrict__ memory, const float* __restrict__ t2v_w0,
    const float* __restrict__ t2v_b0, const float* __restrict__ t2v_w,
    const float* __restrict__ t2v_b, const float* __restrict__ enc_W,
    const float* __restrict__ enc_b, _Float16* __restrict__ xcommon,
    _Float16* __restrict__ s16, _Float16* __restrict__ d16) {
  int e = blockIdx.x;
  for (int c = threadIdx.x; c < 352; c += blockDim.x) {
    if (c < 128) {
      s16[(size_t)e * 128 + c] = (_Float16)memory[(size_t)src_nodes[e] * 128 + c];
    } else if (c < 256) {
      int j = c - 128;
      d16[(size_t)e * 128 + j] = (_Float16)memory[(size_t)dst_nodes[e] * 128 + j];
    } else if (c < 320) {
      int j = c - 256;
      float acc = enc_b[j];
      for (int q = 0; q < 64; ++q)
        acc += edge_attrs[(size_t)e * 64 + q] * enc_W[q * 64 + j];
      xcommon[(size_t)e * 96 + j] = (_Float16)acc;
    } else {
      int j = c - 320;
      float t = timestamps[e];
      float v = (j == 0) ? (t2v_w0[0] * t + t2v_b0[0])
                         : __sinf(t2v_w[j - 1] * t + t2v_b[j - 1]);
      xcommon[(size_t)e * 96 + 64 + j] = (_Float16)v;
    }
  }
}

// ---------------------------------------------------------------- neighbor agg
// grid = (B/2, side, layer). Block: 2 nodes * 20 neighbors = 40 rows (pad to 48).
// nt-major tiling: each wave loads a weight B-fragment once and feeds all 3
// M-tiles (register accumulators) -> 3x less L2 weight traffic.
__global__ __launch_bounds__(256) void neighbor_msg_kernel(
    const int* __restrict__ src_nodes, const int* __restrict__ dst_nodes,
    const float* __restrict__ memory, const int* __restrict__ neighbor_ids,
    const float* __restrict__ neighbor_times,
    const float* __restrict__ neighbor_edge_attrs,
    const float* __restrict__ t2v_w0, const float* __restrict__ t2v_b0,
    const float* __restrict__ t2v_w, const float* __restrict__ t2v_b,
    const _Float16* __restrict__ w1pack_all, const float* __restrict__ b1_all,
    const _Float16* __restrict__ w2pack_all, const float* __restrict__ b2_all,
    float* __restrict__ nb_agg) {
  __shared__ __align__(32) unsigned char smem[48 * IN_DIM * 2 + 48 * HIDDEN * 2];
  _Float16* Xs = (_Float16*)smem;
  _Float16* Hs = (_Float16*)(smem + 48 * IN_DIM * 2);
  float* Msg = (float*)smem;  // aliases Xs; used only after GEMM1 completes

  const int lane = threadIdx.x & 31;
  const int wave = threadIdx.x >> 5;
  const int side = blockIdx.y;
  const int layer = blockIdx.z;
  const int* nodes = side ? dst_nodes : src_nodes;
  const _Float16* w1p = w1pack_all + (size_t)layer * IN_DIM * HIDDEN;
  const _Float16* w2p = w2pack_all + (size_t)layer * HIDDEN * NODE_DIM;
  const float* b1 = b1_all + layer * HIDDEN;
  const float* b2 = b2_all + layer * NODE_DIM;

  // Phase 1: gather + build X tile in f16 (rows 40..47 zero).
  const float w0v = t2v_w0[0], b0v = t2v_b0[0];
  for (int idx = threadIdx.x; idx < 48 * IN_DIM; idx += 256) {
    int r = idx / IN_DIM, c = idx - r * IN_DIM;
    float v = 0.f;
    if (r < 2 * KNBR) {
      int nloc = r / KNBR, k = r - nloc * KNBR;
      int b = blockIdx.x * 2 + nloc;
      int nid = nodes[b];
      if (c < 128) {
        int nbid = neighbor_ids[nid * KNBR + k];
        v = memory[(size_t)nbid * NODE_DIM + c];
      } else if (c < 256) {
        v = memory[(size_t)nid * NODE_DIM + (c - 128)];
      } else if (c < 320) {
        v = neighbor_edge_attrs[((size_t)nid * KNBR + k) * EDGE_DIM + (c - 256)];
      } else {
        int j = c - 320;
        float t = neighbor_times[nid * KNBR + k];
        v = (j == 0) ? (w0v * t + b0v) : __sinf(t2v_w[j - 1] * t + t2v_b[j - 1]);
      }
    }
    Xs[idx] = (_Float16)v;
  }
  __syncthreads();

  // Phase 2: GEMM1 [48x352]@[352x256] + bias + ReLU -> Hs (f16).
  // Wave handles nt0=wave, nt1=wave+8; 3 M-tiles each (6 accumulators).
  {
    const int nt0 = wave, nt1 = wave + 8;
    v8f a00 = {}, a01 = {}, a02 = {}, a10 = {}, a11 = {}, a12 = {};
#pragma unroll
    for (int ks = 0; ks < 11; ++ks) {
      v16h b0 = load_b_pack(w1p, nt0, ks, 11, lane);
      v16h b1 = load_b_pack(w1p, nt1, ks, 11, lane);
      v16h x0 = load_a_rowmajor(Xs, 0, IN_DIM, ks * 32, lane);
      v16h x1 = load_a_rowmajor(Xs, 16, IN_DIM, ks * 32, lane);
      v16h x2 = load_a_rowmajor(Xs, 32, IN_DIM, ks * 32, lane);
      a00 = wmma16x16x32(x0, b0, a00);
      a01 = wmma16x16x32(x1, b0, a01);
      a02 = wmma16x16x32(x2, b0, a02);
      a10 = wmma16x16x32(x0, b1, a10);
      a11 = wmma16x16x32(x1, b1, a11);
      a12 = wmma16x16x32(x2, b1, a12);
    }
    store_tile_relu(Hs, HIDDEN, a00, 0, nt0, b1, lane);
    store_tile_relu(Hs, HIDDEN, a01, 1, nt0, b1, lane);
    store_tile_relu(Hs, HIDDEN, a02, 2, nt0, b1, lane);
    store_tile_relu(Hs, HIDDEN, a10, 0, nt1, b1, lane);
    store_tile_relu(Hs, HIDDEN, a11, 1, nt1, b1, lane);
    store_tile_relu(Hs, HIDDEN, a12, 2, nt1, b1, lane);
  }
  __syncthreads();

  // Phase 3: GEMM2 [48x256]@[256x128] -> Msg (f32, aliases Xs). nt = wave.
  {
    const int nt = wave;
    v8f m0 = {}, m1 = {}, m2 = {};
#pragma unroll
    for (int ks = 0; ks < 8; ++ks) {
      v16h b = load_b_pack(w2p, nt, ks, 8, lane);
      v16h h0 = load_a_rowmajor(Hs, 0, HIDDEN, ks * 32, lane);
      v16h h1 = load_a_rowmajor(Hs, 16, HIDDEN, ks * 32, lane);
      v16h h2 = load_a_rowmajor(Hs, 32, HIDDEN, ks * 32, lane);
      m0 = wmma16x16x32(h0, b, m0);
      m1 = wmma16x16x32(h1, b, m1);
      m2 = wmma16x16x32(h2, b, m2);
    }
    int col = (nt << 4) + (lane & 15);
    int rb = (lane >> 4) << 3;
#pragma unroll
    for (int i = 0; i < 8; ++i) {
      Msg[(rb + i) * NODE_DIM + col] = m0[i];
      Msg[(16 + rb + i) * NODE_DIM + col] = m1[i];
      Msg[(32 + rb + i) * NODE_DIM + col] = m2[i];
    }
  }
  __syncthreads();

  // Phase 4: mean over K + b2 -> nb_agg. 2*128 outputs, 1 per thread.
  {
    int nloc = threadIdx.x >> 7, c = threadIdx.x & 127;
    float s = 0.f;
#pragma unroll
    for (int k = 0; k < KNBR; ++k) s += Msg[(nloc * KNBR + k) * NODE_DIM + c];
    int b = blockIdx.x * 2 + nloc;
    nb_agg[(((size_t)layer * 2 + side) * B_EDGES + b) * NODE_DIM + c] =
        s * (1.f / (float)KNBR) + b2[c];
  }
}

// ---------------------------------------------------------------- edge msg + GRU
// 16 edges/block -> 32 rows (0..15 src-direction, 16..31 dst-direction).
__global__ __launch_bounds__(256) void edge_gru_kernel(
    const _Float16* __restrict__ s16_in, const _Float16* __restrict__ d16_in,
    const _Float16* __restrict__ xcommon, const _Float16* __restrict__ w1p,
    const float* __restrict__ b1, const _Float16* __restrict__ w2p,
    const float* __restrict__ b2, const float* __restrict__ nb_layer,
    const _Float16* __restrict__ wihp, const _Float16* __restrict__ whhp,
    const float* __restrict__ bih, const float* __restrict__ bhh,
    float* __restrict__ s32_out, _Float16* __restrict__ s16_out,
    float* __restrict__ d32_out, _Float16* __restrict__ d16_out) {
  __shared__ __align__(32) _Float16 Xs[32 * IN_DIM];    // keeps h in cols 0..127
  __shared__ __align__(32) _Float16 Hs[32 * HIDDEN];
  __shared__ __align__(32) _Float16 Mf[32 * NODE_DIM];  // GRU input m (f16)
  const int lane = threadIdx.x & 31;
  const int wave = threadIdx.x >> 5;
  const int e0 = blockIdx.x * 16;

  for (int idx = threadIdx.x; idx < 32 * IN_DIM; idx += 256) {
    int r = idx / IN_DIM, c = idx - r * IN_DIM;
    int e = e0 + (r & 15);
    bool sd = r < 16;
    _Float16 v;
    if (c < 128)      v = (sd ? s16_in : d16_in)[(size_t)e * 128 + c];
    else if (c < 256) v = (sd ? d16_in : s16_in)[(size_t)e * 128 + (c - 128)];
    else              v = xcommon[(size_t)e * 96 + (c - 256)];
    Xs[idx] = v;
  }
  __syncthreads();

  // GEMM1 -> Hs (ReLU). Wave: nt0=wave, nt1=wave+8; 2 M-tiles.
  {
    const int nt0 = wave, nt1 = wave + 8;
    v8f a00 = {}, a01 = {}, a10 = {}, a11 = {};
#pragma unroll
    for (int ks = 0; ks < 11; ++ks) {
      v16h b0 = load_b_pack(w1p, nt0, ks, 11, lane);
      v16h b1 = load_b_pack(w1p, nt1, ks, 11, lane);
      v16h x0 = load_a_rowmajor(Xs, 0, IN_DIM, ks * 32, lane);
      v16h x1 = load_a_rowmajor(Xs, 16, IN_DIM, ks * 32, lane);
      a00 = wmma16x16x32(x0, b0, a00);
      a01 = wmma16x16x32(x1, b0, a01);
      a10 = wmma16x16x32(x0, b1, a10);
      a11 = wmma16x16x32(x1, b1, a11);
    }
    store_tile_relu(Hs, HIDDEN, a00, 0, nt0, b1, lane);
    store_tile_relu(Hs, HIDDEN, a01, 1, nt0, b1, lane);
    store_tile_relu(Hs, HIDDEN, a10, 0, nt1, b1, lane);
    store_tile_relu(Hs, HIDDEN, a11, 1, nt1, b1, lane);
  }
  __syncthreads();

  // GEMM2 + b2 + neighbor aggregate -> Mf (f16). nt = wave; 2 M-tiles.
  {
    const int nt = wave;
    v8f m0 = {}, m1 = {};
#pragma unroll
    for (int ks = 0; ks < 8; ++ks) {
      v16h b = load_b_pack(w2p, nt, ks, 8, lane);
      v16h h0 = load_a_rowmajor(Hs, 0, HIDDEN, ks * 32, lane);
      v16h h1 = load_a_rowmajor(Hs, 16, HIDDEN, ks * 32, lane);
      m0 = wmma16x16x32(h0, b, m0);
      m1 = wmma16x16x32(h1, b, m1);
    }
    int col = (nt << 4) + (lane & 15);
    float bias = b2[col];
    int rb = (lane >> 4) << 3;
#pragma unroll
    for (int i = 0; i < 8; ++i) {
      int e = e0 + (rb + i);
      float ms = m0[i] + bias + nb_layer[(size_t)e * NODE_DIM + col];
      float md = m1[i] + bias +
                 nb_layer[((size_t)B_EDGES + e) * NODE_DIM + col];
      Mf[(rb + i) * NODE_DIM + col] = (_Float16)ms;
      Mf[(16 + rb + i) * NODE_DIM + col] = (_Float16)md;
    }
  }
  __syncthreads();

  // GRU: wave handles ct = wave (one 16-col group), both M-tiles.
  // 12 accumulators: gx/gh for gates (r,z,n) x 2 row tiles; B-frags reused 2x.
  {
    const int ct = wave;
    v8f gx[3][2] = {}, gh[3][2] = {};
#pragma unroll
    for (int ks = 0; ks < 4; ++ks) {
      v16h am0 = load_a_rowmajor(Mf, 0, NODE_DIM, ks * 32, lane);
      v16h am1 = load_a_rowmajor(Mf, 16, NODE_DIM, ks * 32, lane);
      v16h ah0 = load_a_rowmajor(Xs, 0, IN_DIM, ks * 32, lane);   // h cols 0..127
      v16h ah1 = load_a_rowmajor(Xs, 16, IN_DIM, ks * 32, lane);
#pragma unroll
      for (int g = 0; g < 3; ++g) {
        v16h bx = load_b_pack(wihp, g * 8 + ct, ks, 4, lane);
        gx[g][0] = wmma16x16x32(am0, bx, gx[g][0]);
        gx[g][1] = wmma16x16x32(am1, bx, gx[g][1]);
        v16h bh = load_b_pack(whhp, g * 8 + ct, ks, 4, lane);
        gh[g][0] = wmma16x16x32(ah0, bh, gh[g][0]);
        gh[g][1] = wmma16x16x32(ah1, bh, gh[g][1]);
      }
    }
    int cl = (ct << 4) + (lane & 15);
    float bxr = bih[cl], bxz = bih[128 + cl], bxn = bih[256 + cl];
    float bhr = bhh[cl], bhz = bhh[128 + cl], bhn = bhh[256 + cl];
#pragma unroll
    for (int mt = 0; mt < 2; ++mt) {
      int rb = (mt << 4) + ((lane >> 4) << 3);
#pragma unroll
      for (int i = 0; i < 8; ++i) {
        int rg = rb + i;
        float r = 1.f / (1.f + __expf(-((gx[0][mt][i] + bxr) + (gh[0][mt][i] + bhr))));
        float z = 1.f / (1.f + __expf(-((gx[1][mt][i] + bxz) + (gh[1][mt][i] + bhz))));
        float n = tanhf((gx[2][mt][i] + bxn) + r * (gh[2][mt][i] + bhn));
        float h = (float)Xs[rg * IN_DIM + cl];
        float hn = (1.f - z) * n + z * h;
        int e = e0 + (rg & 15);
        if (mt == 0) {
          s32_out[(size_t)e * 128 + cl] = hn;
          s16_out[(size_t)e * 128 + cl] = (_Float16)hn;
        } else {
          d32_out[(size_t)e * 128 + cl] = hn;
          d16_out[(size_t)e * 128 + cl] = (_Float16)hn;
        }
      }
    }
  }
}

// ---------------------------------------------------------------- final concat
__global__ void concat_out_kernel(const float* __restrict__ s,
                                  const float* __restrict__ d,
                                  float* __restrict__ out) {
  int id = blockIdx.x * 256 + threadIdx.x;  // B*256 threads
  int e = id >> 8, c = id & 255;
  out[id] = (c < 128) ? s[(size_t)e * 128 + c] : d[(size_t)e * 128 + (c - 128)];
}

// ---------------------------------------------------------------- launch
extern "C" void kernel_launch(void* const* d_in, const int* in_sizes, int n_in,
                              void* d_out, int out_size, void* d_ws,
                              size_t ws_size, hipStream_t stream) {
  (void)in_sizes; (void)n_in; (void)out_size; (void)ws_size;
  const int*   src_nodes = (const int*)d_in[0];
  const int*   dst_nodes = (const int*)d_in[1];
  const float* edge_attrs = (const float*)d_in[2];
  const float* timestamps = (const float*)d_in[3];
  const float* memory = (const float*)d_in[4];
  const int*   neighbor_ids = (const int*)d_in[5];
  const float* neighbor_times = (const float*)d_in[6];
  const float* neighbor_edge_attrs = (const float*)d_in[7];
  const float* t2v_w0 = (const float*)d_in[8];
  const float* t2v_b0 = (const float*)d_in[9];
  const float* t2v_w = (const float*)d_in[10];
  const float* t2v_b = (const float*)d_in[11];
  const float* enc_W = (const float*)d_in[12];
  const float* enc_b = (const float*)d_in[13];
  const float* mlp_W1 = (const float*)d_in[14];
  const float* mlp_b1 = (const float*)d_in[15];
  const float* mlp_W2 = (const float*)d_in[16];
  const float* mlp_b2 = (const float*)d_in[17];
  const float* gru_w_ih = (const float*)d_in[18];
  const float* gru_w_hh = (const float*)d_in[19];
  const float* gru_b_ih = (const float*)d_in[20];
  const float* gru_b_hh = (const float*)d_in[21];

  char* w = (char*)d_ws;
  size_t off = 0;
  auto take = [&](size_t n) -> void* {
    void* p = w + off;
    off += (n + 255) & ~(size_t)255;
    return p;
  };
  _Float16* w1p = (_Float16*)take((size_t)2 * IN_DIM * HIDDEN * 2);
  _Float16* w2p = (_Float16*)take((size_t)2 * HIDDEN * NODE_DIM * 2);
  _Float16* wihp = (_Float16*)take((size_t)NODE_DIM * 384 * 2);
  _Float16* whhp = (_Float16*)take((size_t)NODE_DIM * 384 * 2);
  _Float16* xcommon = (_Float16*)take((size_t)B_EDGES * 96 * 2);
  _Float16* s16a = (_Float16*)take((size_t)B_EDGES * 128 * 2);
  _Float16* s16b = (_Float16*)take((size_t)B_EDGES * 128 * 2);
  _Float16* d16a = (_Float16*)take((size_t)B_EDGES * 128 * 2);
  _Float16* d16b = (_Float16*)take((size_t)B_EDGES * 128 * 2);
  float* s32a = (float*)take((size_t)B_EDGES * 128 * 4);
  float* s32b = (float*)take((size_t)B_EDGES * 128 * 4);
  float* d32a = (float*)take((size_t)B_EDGES * 128 * 4);
  float* d32b = (float*)take((size_t)B_EDGES * 128 * 4);
  float* nb_agg = (float*)take((size_t)2 * 2 * B_EDGES * 128 * 4);

  // Weight prepack (f16, WMMA fragment layout).
  prepack_kernel<<<(IN_DIM * HIDDEN + 255) / 256, 256, 0, stream>>>(
      mlp_W1, w1p, IN_DIM, HIDDEN, 0);
  prepack_kernel<<<(IN_DIM * HIDDEN + 255) / 256, 256, 0, stream>>>(
      mlp_W1 + IN_DIM * HIDDEN, w1p + IN_DIM * HIDDEN, IN_DIM, HIDDEN, 0);
  prepack_kernel<<<(HIDDEN * NODE_DIM + 255) / 256, 256, 0, stream>>>(
      mlp_W2, w2p, HIDDEN, NODE_DIM, 0);
  prepack_kernel<<<(HIDDEN * NODE_DIM + 255) / 256, 256, 0, stream>>>(
      mlp_W2 + HIDDEN * NODE_DIM, w2p + HIDDEN * NODE_DIM, HIDDEN, NODE_DIM, 0);
  prepack_kernel<<<(NODE_DIM * 384 + 255) / 256, 256, 0, stream>>>(
      gru_w_ih, wihp, NODE_DIM, 384, 1);
  prepack_kernel<<<(NODE_DIM * 384 + 255) / 256, 256, 0, stream>>>(
      gru_w_hh, whhp, NODE_DIM, 384, 1);

  // Per-edge init: time2vec + edge encoder + memory gather (f16).
  init_edges_kernel<<<B_EDGES, 256, 0, stream>>>(
      src_nodes, dst_nodes, edge_attrs, timestamps, memory, t2v_w0, t2v_b0,
      t2v_w, t2v_b, enc_W, enc_b, xcommon, s16a, d16a);

  // Neighbor aggregation: all (side, layer) combinations at once.
  dim3 ngrid(B_EDGES / 2, 2, 2);
  neighbor_msg_kernel<<<ngrid, 256, 0, stream>>>(
      src_nodes, dst_nodes, memory, neighbor_ids, neighbor_times,
      neighbor_edge_attrs, t2v_w0, t2v_b0, t2v_w, t2v_b, w1p, mlp_b1, w2p,
      mlp_b2, nb_agg);

  // Layer 0 (a -> b), Layer 1 (b -> final).
  edge_gru_kernel<<<B_EDGES / 16, 256, 0, stream>>>(
      s16a, d16a, xcommon, w1p, mlp_b1, w2p, mlp_b2, nb_agg, wihp, whhp,
      gru_b_ih, gru_b_hh, s32a, s16b, d32a, d16b);
  edge_gru_kernel<<<B_EDGES / 16, 256, 0, stream>>>(
      s16b, d16b, xcommon, w1p + IN_DIM * HIDDEN, mlp_b1 + HIDDEN,
      w2p + HIDDEN * NODE_DIM, mlp_b2 + NODE_DIM,
      nb_agg + (size_t)2 * B_EDGES * 128, wihp, whhp, gru_b_ih, gru_b_hh, s32b,
      s16a, d32b, d16a);

  concat_out_kernel<<<(B_EDGES * 256) / 256, 256, 0, stream>>>(
      s32b, d32b, (float*)d_out);
}
